// MetaLayer_9955734192752
// MI455X (gfx1250) — matrix operand order repaired and checked by reading, not compiled, not run on hardware
//
#include <hip/hip_runtime.h>
#include <hip/hip_bf16.h>

// ---------------------------------------------------------------------------
// MI455X (gfx1250, wave32) MetaLayer GNN.
// GEMMs on v_wmma_f32_16x16x32_f16. Each wave owns TWO 16-row M-tiles
// (32 edges/nodes) so each K-chunk's 4 B fragments are loaded from LDS once
// and reused for 8 WMMAs. Weights staged transposed+f16 in LDS; A fragments
// built in registers from gathered f32 rows (L2-resident; x_a+x_b=51MB fits
// the 192MB L2); segment means via f32 global atomics.
// ---------------------------------------------------------------------------

typedef _Float16 v16h __attribute__((ext_vector_type(16)));
typedef _Float16 v8h  __attribute__((ext_vector_type(8)));
typedef float    v8f  __attribute__((ext_vector_type(8)));
typedef float    v4f  __attribute__((ext_vector_type(4)));

__device__ __forceinline__ int imin(int a, int b) { return a < b ? a : b; }

__device__ __forceinline__ v8f wmma_f16(v16h a, v16h b, v8f c) {
  // (neg_a, A, neg_b, B, c_mod, C, reuse_a, reuse_b)
  return __builtin_amdgcn_wmma_f32_16x16x32_f16(false, a, false, b, (short)0, c,
                                                false, false);
}

// Build a 16x32 f16 A-fragment slice for this lane from an f32 row.
// ISA 7.12.2 (16-bit A 16x32): lanes 0-15 hold K = base+[0..7] and
// base+16+[0..7] with base=0; lanes 16-31 same with base=8.
// `off` = within-row float offset of K=base; scaled by `s` before cvt.
__device__ __forceinline__ v16h load_a_row(const float* __restrict__ row,
                                           int off, float s) {
  const v4f* p = (const v4f*)(row + off);
  const v4f* q = (const v4f*)(row + off + 16);
  v4f f0 = p[0], f1 = p[1], f2 = q[0], f3 = q[1];
  v16h a;
  a[0]  = (_Float16)(f0[0] * s); a[1]  = (_Float16)(f0[1] * s);
  a[2]  = (_Float16)(f0[2] * s); a[3]  = (_Float16)(f0[3] * s);
  a[4]  = (_Float16)(f1[0] * s); a[5]  = (_Float16)(f1[1] * s);
  a[6]  = (_Float16)(f1[2] * s); a[7]  = (_Float16)(f1[3] * s);
  a[8]  = (_Float16)(f2[0] * s); a[9]  = (_Float16)(f2[1] * s);
  a[10] = (_Float16)(f2[2] * s); a[11] = (_Float16)(f2[3] * s);
  a[12] = (_Float16)(f3[0] * s); a[13] = (_Float16)(f3[1] * s);
  a[14] = (_Float16)(f3[2] * s); a[15] = (_Float16)(f3[3] * s);
  return a;
}

// B fragment: per-lane fixed column N, K = koff+[0..7] and koff+16+[0..7] from
// LDS-resident transposed weights WT[n*stride + k]. 16B aligned -> 2x ds_load_b128.
__device__ __forceinline__ v16h load_b(const _Float16* __restrict__ wt,
                                       int stride, int n, int koff) {
  v8h lo = *(const v8h*)(wt + n * stride + koff);
  v8h hh = *(const v8h*)(wt + n * stride + koff + 16);
  return __builtin_shufflevector(lo, hh, 0, 1, 2, 3, 4, 5, 6, 7, 8, 9, 10, 11,
                                 12, 13, 14, 15);
}

// A fragment for GEMM2 from LDS hidden (row-major f16, stride DH).
__device__ __forceinline__ v16h load_a_hid(const _Float16* __restrict__ hid,
                                           int m, int stride, int off) {
  v8h lo = *(const v8h*)(hid + m * stride + off);
  v8h hh = *(const v8h*)(hid + m * stride + off + 16);
  return __builtin_shufflevector(lo, hh, 0, 1, 2, 3, 4, 5, 6, 7, 8, 9, 10, 11,
                                 12, 13, 14, 15);
}

// ---------------------------------------------------------------------------
// Edge MLP: ne = relu(concat(xs[src], xd[dst], ea) @ We1 + be1) @ We2 + be2
// Block = 256 threads = 8 waves; 32 edges/wave (2 M-tiles), 256 edges/block.
// ---------------------------------------------------------------------------
__global__ __launch_bounds__(256)
void edge_mlp_kernel(const float* __restrict__ xs, const float* __restrict__ xd,
                     const int* __restrict__ ei, const float* __restrict__ ea,
                     const float* __restrict__ We1, const float* __restrict__ be1,
                     const float* __restrict__ We2, const float* __restrict__ be2,
                     float* __restrict__ ne, int E) {
  constexpr int DN = 64, DE = 32, DH = 64, KIN = 2 * DN + DE;  // 160
  __shared__ _Float16 WT1[DH * KIN];       // We1^T [n][k], f16 (20KB)
  __shared__ _Float16 WT2[DE * DH];        // We2^T [n][k], f16 (4KB)
  __shared__ _Float16 HID[8 * 32 * DH];    // per-wave hidden 32x64 f16 (32KB)

  const int tid = threadIdx.x;
  for (int i = tid; i < DH * KIN; i += 256) {
    int n = i / KIN, k = i - n * KIN;
    WT1[i] = (_Float16)We1[k * DH + n];
  }
  for (int i = tid; i < DE * DH; i += 256) {
    int n = i / DH, k = i - n * DH;
    WT2[i] = (_Float16)We2[k * DE + n];
  }
  __syncthreads();

  const int wave = tid >> 5, lane = tid & 31;
  const int m = lane & 15, hi = lane >> 4, base = hi << 3;
  const int e0 = blockIdx.x * 256 + wave * 32;

  // Per-lane gathered rows for the two M-tiles (clamped; EXEC stays all-1s).
  const float* srow[2];
  const float* drow[2];
  const float* erow[2];
#pragma unroll
  for (int t = 0; t < 2; ++t) {
    const int e = imin(e0 + t * 16 + m, E - 1);
    srow[t] = xs + (size_t)ei[e] * DN;
    drow[t] = xd + (size_t)ei[E + e] * DN;
    erow[t] = ea + (size_t)e * DE;
  }
  // Prefetch next block's (sequential) edge-attribute stream one block ahead.
  __builtin_prefetch(erow[0] + 256 * DE, 0, 1);

  // GEMM1: [32x160] x [160x64]; 5 K-chunks, B fragments reused across 2 tiles.
  v8f acc[2][4];
#pragma unroll
  for (int t = 0; t < 2; ++t)
#pragma unroll
    for (int nt = 0; nt < 4; ++nt) acc[t][nt] = 0;

#pragma unroll
  for (int c = 0; c < 5; ++c) {
    const int koff = c * 32 + hi * 8;
    v16h b0 = load_b(WT1, KIN, 0 * 16 + m, koff);
    v16h b1 = load_b(WT1, KIN, 1 * 16 + m, koff);
    v16h b2 = load_b(WT1, KIN, 2 * 16 + m, koff);
    v16h b3 = load_b(WT1, KIN, 3 * 16 + m, koff);
#pragma unroll
    for (int t = 0; t < 2; ++t) {
      v16h a;
      if (c < 2)      a = load_a_row(srow[t], c * 32 + base, 1.0f);
      else if (c < 4) a = load_a_row(drow[t], (c - 2) * 32 + base, 1.0f);
      else            a = load_a_row(erow[t], base, 1.0f);
      acc[t][0] = wmma_f16(a, b0, acc[t][0]);
      acc[t][1] = wmma_f16(a, b1, acc[t][1]);
      acc[t][2] = wmma_f16(a, b2, acc[t][2]);
      acc[t][3] = wmma_f16(a, b3, acc[t][3]);
    }
  }

  // Bias + ReLU; re-layout D(C-layout) -> A-layout through per-wave LDS.
  _Float16* hid = HID + wave * 32 * DH;
#pragma unroll
  for (int nt = 0; nt < 4; ++nt) {
    const int n = nt * 16 + m;
    const float bias = be1[n];
#pragma unroll
    for (int t = 0; t < 2; ++t)
#pragma unroll
      for (int v = 0; v < 8; ++v) {
        const int mm = t * 16 + base + v;  // C layout: lanes>=16 hold M=8..15
        hid[mm * DH + n] = (_Float16)fmaxf(acc[t][nt][v] + bias, 0.0f);
      }
  }
  // Same-wave DS ops are in-order; no barrier required.

  // GEMM2: [32x64] x [64x32]; 2 K-chunks x 2 N-tiles, B reused across tiles.
  v8f oacc[2][2];
#pragma unroll
  for (int t = 0; t < 2; ++t) { oacc[t][0] = 0; oacc[t][1] = 0; }
#pragma unroll
  for (int c = 0; c < 2; ++c) {
    const int koff = c * 32 + hi * 8;
    v16h b0 = load_b(WT2, DH, 0 * 16 + m, koff);
    v16h b1 = load_b(WT2, DH, 1 * 16 + m, koff);
#pragma unroll
    for (int t = 0; t < 2; ++t) {
      v16h a = load_a_hid(hid, t * 16 + m, DH, c * 32 + base);
      oacc[t][0] = wmma_f16(a, b0, oacc[t][0]);
      oacc[t][1] = wmma_f16(a, b1, oacc[t][1]);
    }
  }

#pragma unroll
  for (int nt = 0; nt < 2; ++nt) {
    const int n = nt * 16 + m;
    const float bias = be2[n];
#pragma unroll
    for (int t = 0; t < 2; ++t)
#pragma unroll
      for (int v = 0; v < 8; ++v) {
        const int ee = e0 + t * 16 + base + v;
        if (ee < E) ne[(size_t)ee * DE + n] = oacc[t][nt][v] + bias;
      }
  }
}

// ---------------------------------------------------------------------------
// Node MLP: nx = relu(concat(x, sum1/cnt1, sum2/cnt2) @ Wn1 + bn1) @ Wn2 + bn2
// Block = 256 threads = 8 waves; 32 nodes/wave (2 M-tiles), 256 nodes/block.
// ---------------------------------------------------------------------------
__global__ __launch_bounds__(256)
void node_mlp_kernel(const float* __restrict__ x,
                     const float* __restrict__ sum1, const float* __restrict__ cnt1,
                     const float* __restrict__ sum2, const float* __restrict__ cnt2,
                     const float* __restrict__ Wn1, const float* __restrict__ bn1,
                     const float* __restrict__ Wn2, const float* __restrict__ bn2,
                     float* __restrict__ nx, int N) {
  constexpr int DN = 64, DE = 32, DH = 64, KIN = DN + 2 * DE;  // 128
  __shared__ _Float16 WT1[DH * KIN];     // Wn1^T [n][k] (16KB)
  __shared__ _Float16 WT2[DN * DH];      // Wn2^T [n][k] (8KB)
  __shared__ _Float16 HID[8 * 32 * DH];  // per-wave hidden (32KB)

  const int tid = threadIdx.x;
  for (int i = tid; i < DH * KIN; i += 256) {
    int n = i / KIN, k = i - n * KIN;
    WT1[i] = (_Float16)Wn1[k * DH + n];
  }
  for (int i = tid; i < DN * DH; i += 256) {
    int n = i / DH, k = i - n * DH;
    WT2[i] = (_Float16)Wn2[k * DN + n];
  }
  __syncthreads();

  const int wave = tid >> 5, lane = tid & 31;
  const int m = lane & 15, hi = lane >> 4, base = hi << 3;
  const int r0 = blockIdx.x * 256 + wave * 32;

  const float* xrow[2];
  const float* s1row[2];
  const float* s2row[2];
  float inv1[2], inv2[2];
#pragma unroll
  for (int t = 0; t < 2; ++t) {
    const int r = imin(r0 + t * 16 + m, N - 1);
    xrow[t]  = x    + (size_t)r * DN;
    s1row[t] = sum1 + (size_t)r * DE;
    s2row[t] = sum2 + (size_t)r * DE;
    inv1[t] = 1.0f / fmaxf(cnt1[r], 1.0f);
    inv2[t] = 1.0f / fmaxf(cnt2[r], 1.0f);
  }

  // GEMM1: [32x128] x [128x64]; 4 K-chunks x 4 N-tiles, B reused across tiles.
  v8f acc[2][4];
#pragma unroll
  for (int t = 0; t < 2; ++t)
#pragma unroll
    for (int nt = 0; nt < 4; ++nt) acc[t][nt] = 0;

#pragma unroll
  for (int c = 0; c < 4; ++c) {
    const int koff = c * 32 + hi * 8;
    v16h b0 = load_b(WT1, KIN, 0 * 16 + m, koff);
    v16h b1 = load_b(WT1, KIN, 1 * 16 + m, koff);
    v16h b2 = load_b(WT1, KIN, 2 * 16 + m, koff);
    v16h b3 = load_b(WT1, KIN, 3 * 16 + m, koff);
#pragma unroll
    for (int t = 0; t < 2; ++t) {
      v16h a;
      if (c < 2)       a = load_a_row(xrow[t], c * 32 + base, 1.0f);
      else if (c == 2) a = load_a_row(s1row[t], base, inv1[t]);
      else             a = load_a_row(s2row[t], base, inv2[t]);
      acc[t][0] = wmma_f16(a, b0, acc[t][0]);
      acc[t][1] = wmma_f16(a, b1, acc[t][1]);
      acc[t][2] = wmma_f16(a, b2, acc[t][2]);
      acc[t][3] = wmma_f16(a, b3, acc[t][3]);
    }
  }

  _Float16* hid = HID + wave * 32 * DH;
#pragma unroll
  for (int nt = 0; nt < 4; ++nt) {
    const int n = nt * 16 + m;
    const float bias = bn1[n];
#pragma unroll
    for (int t = 0; t < 2; ++t)
#pragma unroll
      for (int v = 0; v < 8; ++v) {
        const int mm = t * 16 + base + v;
        hid[mm * DH + n] = (_Float16)fmaxf(acc[t][nt][v] + bias, 0.0f);
      }
  }

  // GEMM2: [32x64] x [64x64]; 2 K-chunks x 4 N-tiles, B reused across tiles.
  v8f oacc[2][4];
#pragma unroll
  for (int t = 0; t < 2; ++t)
#pragma unroll
    for (int nt = 0; nt < 4; ++nt) oacc[t][nt] = 0;
#pragma unroll
  for (int c = 0; c < 2; ++c) {
    const int koff = c * 32 + hi * 8;
    v16h b0 = load_b(WT2, DH, 0 * 16 + m, koff);
    v16h b1 = load_b(WT2, DH, 1 * 16 + m, koff);
    v16h b2 = load_b(WT2, DH, 2 * 16 + m, koff);
    v16h b3 = load_b(WT2, DH, 3 * 16 + m, koff);
#pragma unroll
    for (int t = 0; t < 2; ++t) {
      v16h a = load_a_hid(hid, t * 16 + m, DH, c * 32 + base);
      oacc[t][0] = wmma_f16(a, b0, oacc[t][0]);
      oacc[t][1] = wmma_f16(a, b1, oacc[t][1]);
      oacc[t][2] = wmma_f16(a, b2, oacc[t][2]);
      oacc[t][3] = wmma_f16(a, b3, oacc[t][3]);
    }
  }

#pragma unroll
  for (int nt = 0; nt < 4; ++nt) {
    const int n = nt * 16 + m;
    const float bias = bn2[n];
#pragma unroll
    for (int t = 0; t < 2; ++t)
#pragma unroll
      for (int v = 0; v < 8; ++v) {
        const int rr = r0 + t * 16 + base + v;
        if (rr < N) nx[(size_t)rr * DN + n] = oacc[t][nt][v] + bias;
      }
  }
}

// ---------------------------------------------------------------------------
// Segment-sum scatter: 8 threads per edge, 4 floats each + one count.
// ---------------------------------------------------------------------------
__global__ __launch_bounds__(256)
void scatter_add_kernel(const float* __restrict__ ne, const int* __restrict__ idx,
                        float* __restrict__ sum, float* __restrict__ cnt, int E) {
  const int t = blockIdx.x * blockDim.x + threadIdx.x;
  const int e = t >> 3;
  if (e >= E) return;
  const int c = (t & 7) * 4;
  const int r = idx[e];
  const float* src = ne + (size_t)e * 32 + c;
  float* dst = sum + (size_t)r * 32 + c;
#pragma unroll
  for (int j = 0; j < 4; ++j) atomicAdd(&dst[j], src[j]);
  if ((t & 7) == 0) atomicAdd(&cnt[r], 1.0f);
}

__global__ __launch_bounds__(256)
void zero_kernel(float* __restrict__ p, long long n) {
  long long i = (long long)blockIdx.x * blockDim.x + threadIdx.x;
  long long stride = (long long)gridDim.x * blockDim.x;
  for (; i < n; i += stride) p[i] = 0.0f;
}

// ---------------------------------------------------------------------------
extern "C" void kernel_launch(void* const* d_in, const int* in_sizes, int n_in,
                              void* d_out, int out_size, void* d_ws, size_t ws_size,
                              hipStream_t stream) {
  constexpr int DN = 64, DE = 32;
  const float* x_a  = (const float*)d_in[0];
  const float* x_b  = (const float*)d_in[1];
  const int* ei_aa  = (const int*)d_in[2];
  const int* ei_ab  = (const int*)d_in[3];
  const int* ei_bb  = (const int*)d_in[4];
  const float* ea_aa = (const float*)d_in[5];
  const float* ea_ab = (const float*)d_in[6];
  const float* ea_bb = (const float*)d_in[7];
  const float* We1 = (const float*)d_in[8];
  const float* be1 = (const float*)d_in[9];
  const float* We2 = (const float*)d_in[10];
  const float* be2 = (const float*)d_in[11];
  const float* Wn1 = (const float*)d_in[12];
  const float* bn1 = (const float*)d_in[13];
  const float* Wn2 = (const float*)d_in[14];
  const float* bn2 = (const float*)d_in[15];

  const int NA = in_sizes[0] / DN;
  const int NB = in_sizes[1] / DN;
  const int E  = in_sizes[2] / 2;

  // Outputs concatenated flat in return order: nx_a, nx_b, ne_aa, ne_ab, ne_bb.
  float* out   = (float*)d_out;
  float* nx_a  = out;
  float* nx_b  = nx_a + (size_t)NA * DN;
  float* ne_aa = nx_b + (size_t)NB * DN;
  float* ne_ab = ne_aa + (size_t)E * DE;
  float* ne_bb = ne_ab + (size_t)E * DE;

  // Workspace: 4 segment sums (N x 32) + 4 counts.
  float* ws    = (float*)d_ws;
  float* sumA1 = ws;
  float* sumA2 = sumA1 + (size_t)NA * DE;
  float* sumB1 = sumA2 + (size_t)NA * DE;
  float* sumB2 = sumB1 + (size_t)NB * DE;
  float* cntA1 = sumB2 + (size_t)NB * DE;
  float* cntA2 = cntA1 + NA;
  float* cntB1 = cntA2 + NA;
  float* cntB2 = cntB1 + NB;
  const long long wsFloats = 2LL * (NA + NB) * DE + 2LL * (NA + NB);

  zero_kernel<<<2048, 256, 0, stream>>>(ws, wsFloats);

  // Edge MLPs (per-type weight slices). 256 edges per block.
  const int KE = 2 * DN + DE;  // 160
  const int egrid = (E + 255) / 256;
  edge_mlp_kernel<<<egrid, 256, 0, stream>>>(
      x_a, x_a, ei_aa, ea_aa, We1 + 0 * KE * DN, be1 + 0 * DN,
      We2 + 0 * DN * DE, be2 + 0 * DE, ne_aa, E);
  edge_mlp_kernel<<<egrid, 256, 0, stream>>>(
      x_a, x_b, ei_ab, ea_ab, We1 + 1 * KE * DN, be1 + 1 * DN,
      We2 + 1 * DN * DE, be2 + 1 * DE, ne_ab, E);
  edge_mlp_kernel<<<egrid, 256, 0, stream>>>(
      x_b, x_b, ei_bb, ea_bb, We1 + 2 * KE * DN, be1 + 2 * DN,
      We2 + 2 * DN * DE, be2 + 2 * DE, ne_bb, E);

  // Segment sums (stream-ordered after edge kernels).
  const int sgrid = (E * 8 + 255) / 256;
  scatter_add_kernel<<<sgrid, 256, 0, stream>>>(ne_aa, ei_aa,     sumA1, cntA1, E);
  scatter_add_kernel<<<sgrid, 256, 0, stream>>>(ne_ab, ei_ab,     sumA2, cntA2, E);
  scatter_add_kernel<<<sgrid, 256, 0, stream>>>(ne_ab, ei_ab + E, sumB1, cntB1, E);
  scatter_add_kernel<<<sgrid, 256, 0, stream>>>(ne_bb, ei_bb,     sumB2, cntB2, E);

  // Node MLPs (per-type weight slices). 256 nodes per block.
  const int KNn = DN + 2 * DE;  // 128
  node_mlp_kernel<<<(NA + 255) / 256, 256, 0, stream>>>(
      x_a, sumA1, cntA1, sumA2, cntA2, Wn1 + 0 * KNn * DN, bn1 + 0 * DN,
      Wn2 + 0 * DN * DN, bn2 + 0 * DN, nx_a, NA);
  node_mlp_kernel<<<(NB + 255) / 256, 256, 0, stream>>>(
      x_b, sumB1, cntB1, sumB2, cntB2, Wn1 + 1 * KNn * DN, bn1 + 1 * DN,
      Wn2 + 1 * DN * DN, bn2 + 1 * DN, nx_b, NB);
}
